// OFTLinear_47132971106962
// MI455X (gfx1250) — matrix operand order repaired and checked by reading, not compiled, not run on hardware
//
#include <hip/hip_runtime.h>

typedef __bf16 bf16x16 __attribute__((ext_vector_type(16)));
typedef float  f32x8   __attribute__((ext_vector_type(8)));
typedef unsigned int u32x4 __attribute__((ext_vector_type(4)));

#define IN_F   4096
#define OUT_F  4096
#define BSZ    64
#define NTRI   2016
#define TOKENS 8192

__device__ __forceinline__ unsigned short f2bf(float f) {
    unsigned int u = __float_as_uint(f);
    u += 0x7FFFu + ((u >> 16) & 1u);       // round-to-nearest-even
    return (unsigned short)(u >> 16);
}

// ---------------------------------------------------------------------------
// invpo[perm_out[k]] = k  (lets the GEMM epilogue scatter instead of gather)
// ---------------------------------------------------------------------------
__global__ void invperm_kernel(const int* __restrict__ perm_out,
                               int* __restrict__ invpo) {
    int k = blockIdx.x * 256 + threadIdx.x;
    invpo[perm_out[k]] = k;
}

// ---------------------------------------------------------------------------
// Build R = I + 2(Q + Q^2 + Q^3 + Q^4) for 128 blocks (64 out, 64 in).
// One workgroup per 64x64 block; powers via LDS ping-pong.
// R_all layout: [blk][64][64], blk 0..63 = R_out, 64..127 = R_in.
// ---------------------------------------------------------------------------
__global__ void build_R_kernel(const float* __restrict__ Ro,
                               const float* __restrict__ Ri,
                               float* __restrict__ R_all) {
    __shared__ float Q[BSZ][BSZ];
    __shared__ float P0[BSZ][BSZ];
    __shared__ float P1[BSZ][BSZ];
    const int blk = blockIdx.x;
    const float* vec = (blk < 64) ? (Ro + blk * NTRI) : (Ri + (blk - 64) * NTRI);
    const int tid = threadIdx.x;

    float acc[16];
#pragma unroll
    for (int e = 0; e < 16; ++e) {
        int t = tid + 256 * e;
        int i = t >> 6, j = t & 63;
        float v = 0.f;
        if (i < j)      v =  vec[i * 63 - (i * (i - 1)) / 2 + (j - i - 1)];
        else if (i > j) v = -vec[j * 63 - (j * (j - 1)) / 2 + (i - j - 1)];
        Q[i][j]  = v;
        P0[i][j] = v;
        acc[e] = v;
    }
    __syncthreads();

    for (int p = 2; p <= 4; ++p) {
        float (*src)[BSZ] = (p & 1) ? P1 : P0;
        float (*dst)[BSZ] = (p & 1) ? P0 : P1;
        float nv[16];
#pragma unroll
        for (int e = 0; e < 16; ++e) {
            int t = tid + 256 * e;
            int i = t >> 6, j = t & 63;
            float s = 0.f;
#pragma unroll 8
            for (int l = 0; l < 64; ++l) s += src[i][l] * Q[l][j];
            nv[e] = s;
        }
#pragma unroll
        for (int e = 0; e < 16; ++e) {
            int t = tid + 256 * e;
            dst[t >> 6][t & 63] = nv[e];
            acc[e] += nv[e];
        }
        __syncthreads();
    }

    float* out = R_all + (size_t)blk * 4096;
#pragma unroll
    for (int e = 0; e < 16; ++e) {
        int t = tid + 256 * e;
        int i = t >> 6, j = t & 63;
        out[t] = ((i == j) ? 1.f : 0.f) + 2.f * acc[e];
    }
}

// ---------------------------------------------------------------------------
// bias2[o] = sum_d bias[ro*64+d] * R_out[ro][d][o%64]
// ---------------------------------------------------------------------------
__global__ void bias2_kernel(const float* __restrict__ bias,
                             const float* __restrict__ R_all,
                             float* __restrict__ bias2) {
    int o  = blockIdx.x * 256 + threadIdx.x;
    int ro = o >> 6, dp = o & 63;
    const float* R = R_all + (size_t)ro * 4096;
    float s = 0.f;
#pragma unroll 8
    for (int d = 0; d < 64; ++d) s += bias[ro * 64 + d] * R[d * 64 + dp];
    bias2[o] = s;
}

// ---------------------------------------------------------------------------
// Fold both rotations into the weight:
//   W3_tile(ro,ri) = R_out[ro]^T * W_blk(ro,ri) * R_in[ri]^T   -> bf16
// One workgroup per 64x64 tile of the 4096x4096 weight.
// ---------------------------------------------------------------------------
__global__ void build_W3_kernel(const float* __restrict__ W,
                                const float* __restrict__ R_all,
                                unsigned short* __restrict__ W3) {
    __shared__ float Wb[BSZ][BSZ];
    __shared__ float Ao[BSZ][BSZ];   // R_out[ro]
    __shared__ float Bi[BSZ][BSZ];   // R_in[ri]
    const int ro = blockIdx.y, ri = blockIdx.x;
    const int tid = threadIdx.x;
    const float* Wblk = W + (size_t)ro * 64 * IN_F + (size_t)ri * 64;
    const float* RoB  = R_all + (size_t)ro * 4096;
    const float* RiB  = R_all + (size_t)(64 + ri) * 4096;

#pragma unroll
    for (int e = 0; e < 16; ++e) {
        int t = tid + 256 * e;
        int i = t >> 6, j = t & 63;
        Wb[i][j] = Wblk[(size_t)i * IN_F + j];
        Ao[i][j] = RoB[t];
        Bi[i][j] = RiB[t];
    }
    __syncthreads();

    // T1 = R_out^T * Wb
    float t1[16];
#pragma unroll
    for (int e = 0; e < 16; ++e) {
        int t = tid + 256 * e;
        int dp = t >> 6, ee = t & 63;
        float s = 0.f;
#pragma unroll 8
        for (int d = 0; d < 64; ++d) s += Ao[d][dp] * Wb[d][ee];
        t1[e] = s;
    }
    __syncthreads();
#pragma unroll
    for (int e = 0; e < 16; ++e) {
        int t = tid + 256 * e;
        Wb[t >> 6][t & 63] = t1[e];
    }
    __syncthreads();

    // T2 = T1 * R_in^T  -> bf16 store, row-major [o][k]
#pragma unroll
    for (int e = 0; e < 16; ++e) {
        int t = tid + 256 * e;
        int dp = t >> 6, c = t & 63;
        float s = 0.f;
#pragma unroll 8
        for (int ee = 0; ee < 64; ++ee) s += Wb[dp][ee] * Bi[c][ee];
        W3[(size_t)(ro * 64 + dp) * IN_F + ri * 64 + c] = f2bf(s);
    }
}

// ---------------------------------------------------------------------------
// XP[m][j] = bf16( x[m][ perm_in_inv[j] ] )  (input permutation + convert)
// ---------------------------------------------------------------------------
__global__ void permx_kernel(const float* __restrict__ x,
                             const int* __restrict__ pii,
                             unsigned short* __restrict__ XP) {
    int idx = blockIdx.x * 256 + threadIdx.x;
    int m = idx >> 12, j = idx & 4095;
    XP[idx] = f2bf(x[((size_t)m << 12) + pii[j]]);
}

// ---------------------------------------------------------------------------
// Main GEMM: out[m, invpo[o]] = sum_k XP[m,k]*W3[o,k] + bias2[o]
// Workgroup tile 128x256 / 256 threads (8 waves, 2x4 wave grid); each wave
// owns 64x64 = 4x4 WMMA tiles -> 16 v_wmma_f32_16x16x32_bf16 per K-step,
// 1 ds_load_b128 per WMMA. Double-buffered LDS: ONE barrier per K-step.
// ---------------------------------------------------------------------------
__global__ __launch_bounds__(256, 1)
void oft_gemm_kernel(const unsigned short* __restrict__ XP,
                     const unsigned short* __restrict__ W3,
                     const float* __restrict__ bias2,
                     const int* __restrict__ invpo,
                     float* __restrict__ out) {
    __shared__ unsigned short lA[2][128][40];   // [buf][m][k], 8-elem row pad
    __shared__ unsigned short lB[2][256][40];   // [buf][n][k]

    const int m0   = blockIdx.y * 128;
    const int n0   = blockIdx.x * 256;
    const int tid  = threadIdx.x;
    const int lane = tid & 31;
    const int wave = tid >> 5;
    const int wm   = wave & 1;    // 2 wave-rows * 64 M
    const int wn   = wave >> 1;   // 4 wave-cols * 64 N
    const int hx   = lane >> 4;   // lane half
    const int nl   = lane & 15;

    // global->LDS staging: A = 2 x 16B, B = 4 x 16B per thread per K-step
    const int gr = tid >> 2;           // 0..63
    const int gc = (tid & 3) << 3;     // 0,8,16,24
    const unsigned short* gA = XP + (size_t)(m0 + gr) * IN_F + gc;
    const unsigned short* gB = W3 + (size_t)(n0 + gr) * IN_F + gc;
    const size_t rs64 = (size_t)64 * IN_F;

    const f32x8 zero = {0.f, 0.f, 0.f, 0.f, 0.f, 0.f, 0.f, 0.f};
    f32x8 acc[4][4];
#pragma unroll
    for (int mt = 0; mt < 4; ++mt)
#pragma unroll
        for (int nt = 0; nt < 4; ++nt) acc[mt][nt] = zero;

    u32x4 ra0, ra1, rb0, rb1, rb2, rb3;
    ra0 = *(const u32x4*)(gA);
    ra1 = *(const u32x4*)(gA + rs64);
    rb0 = *(const u32x4*)(gB);
    rb1 = *(const u32x4*)(gB + rs64);
    rb2 = *(const u32x4*)(gB + 2 * rs64);
    rb3 = *(const u32x4*)(gB + 3 * rs64);
    *(u32x4*)&lA[0][gr][gc]       = ra0;
    *(u32x4*)&lA[0][gr + 64][gc]  = ra1;
    *(u32x4*)&lB[0][gr][gc]       = rb0;
    *(u32x4*)&lB[0][gr + 64][gc]  = rb1;
    *(u32x4*)&lB[0][gr + 128][gc] = rb2;
    *(u32x4*)&lB[0][gr + 192][gc] = rb3;
    __syncthreads();

    int buf = 0;
    for (int k0 = 0; k0 < IN_F; k0 += 32) {
        const bool has_next = (k0 + 32) < IN_F;
        if (has_next) {
            const unsigned short* nA = gA + k0 + 32;
            const unsigned short* nB = gB + k0 + 32;
            ra0 = *(const u32x4*)(nA);
            ra1 = *(const u32x4*)(nA + rs64);
            rb0 = *(const u32x4*)(nB);
            rb1 = *(const u32x4*)(nB + rs64);
            rb2 = *(const u32x4*)(nB + 2 * rs64);
            rb3 = *(const u32x4*)(nB + 3 * rs64);
            if (k0 + 64 < IN_F) {               // one more tile ahead -> L2
                __builtin_prefetch(nA + 32, 0, 1);
                __builtin_prefetch(nB + 32, 0, 1);
            }
        }

        // A frag: lanes 0-15 hold K{0-7,16-23}; lanes 16-31 hold K{8-15,24-31}
        union { bf16x16 v; u32x4 q[2]; } afrag[4], bfrag[4];
#pragma unroll
        for (int mt = 0; mt < 4; ++mt) {
            int row = wm * 64 + mt * 16 + nl;
            afrag[mt].q[0] = *(const u32x4*)&lA[buf][row][hx * 8];
            afrag[mt].q[1] = *(const u32x4*)&lA[buf][row][16 + hx * 8];
        }
        // B frag: lanes 0-15 hold K{0-7,16-23}; lanes 16-31 hold K{8-15,24-31}, N = lane&15
#pragma unroll
        for (int nt = 0; nt < 4; ++nt) {
            int col = wn * 64 + nt * 16 + nl;
            bfrag[nt].q[0] = *(const u32x4*)&lB[buf][col][hx * 8];
            bfrag[nt].q[1] = *(const u32x4*)&lB[buf][col][16 + hx * 8];
        }

#pragma unroll
        for (int mt = 0; mt < 4; ++mt)
#pragma unroll
            for (int nt = 0; nt < 4; ++nt)
                acc[mt][nt] = __builtin_amdgcn_wmma_f32_16x16x32_bf16(
                    false, afrag[mt].v, false, bfrag[nt].v,
                    (short)0, acc[mt][nt], false, false);

        if (has_next) {
            int nb = buf ^ 1;
            *(u32x4*)&lA[nb][gr][gc]       = ra0;
            *(u32x4*)&lA[nb][gr + 64][gc]  = ra1;
            *(u32x4*)&lB[nb][gr][gc]       = rb0;
            *(u32x4*)&lB[nb][gr + 64][gc]  = rb1;
            *(u32x4*)&lB[nb][gr + 128][gc] = rb2;
            *(u32x4*)&lB[nb][gr + 192][gc] = rb3;
            __syncthreads();    // publishes nb; proves all waves done reading buf
            buf = nb;
        }
    }

    // Epilogue: + rotated bias, scatter through inverse output permutation.
    // C layout: VGPR v -> rows v (lanes 0-15) / v+8 (lanes 16-31), N = lane&15.
#pragma unroll
    for (int nt = 0; nt < 4; ++nt) {
        int o   = n0 + wn * 64 + nt * 16 + nl;
        float bo = bias2[o];
        int ko  = invpo[o];
#pragma unroll
        for (int mt = 0; mt < 4; ++mt) {
            int mbase = m0 + wm * 64 + mt * 16 + hx * 8;
            f32x8 a = acc[mt][nt];
#pragma unroll
            for (int v = 0; v < 8; ++v)
                out[(size_t)(mbase + v) * OUT_F + ko] = a[v] + bo;
        }
    }
}

// ---------------------------------------------------------------------------
extern "C" void kernel_launch(void* const* d_in, const int* in_sizes, int n_in,
                              void* d_out, int out_size, void* d_ws, size_t ws_size,
                              hipStream_t stream) {
    const float* x    = (const float*)d_in[0];
    const float* Ro   = (const float*)d_in[1];
    const float* Ri   = (const float*)d_in[2];
    const float* W    = (const float*)d_in[3];
    const float* bias = (const float*)d_in[4];
    const int*   pii  = (const int*)d_in[5];
    const int*   po   = (const int*)d_in[6];
    float* out = (float*)d_out;

    // workspace carve-up
    char* ws = (char*)d_ws;
    float*          R_all = (float*)ws;                                   //  2 MB
    unsigned short* W3    = (unsigned short*)(ws + (2u << 20));           // 32 MB
    unsigned short* XP    = (unsigned short*)(ws + (2u << 20) + (32u << 20)); // 64 MB
    float*          b2    = (float*)(ws + (2u << 20) + (32u << 20) + (64u << 20));
    int*            invpo = (int*)((char*)b2 + 16384);

    invperm_kernel<<<16, 256, 0, stream>>>(po, invpo);
    build_R_kernel<<<128, 256, 0, stream>>>(Ro, Ri, R_all);
    bias2_kernel<<<16, 256, 0, stream>>>(bias, R_all, b2);
    build_W3_kernel<<<dim3(64, 64), 256, 0, stream>>>(W, R_all, W3);
    permx_kernel<<<(TOKENS * IN_F) / 256, 256, 0, stream>>>(x, pii, XP);
    oft_gemm_kernel<<<dim3(OUT_F / 256, TOKENS / 128), 256, 0, stream>>>(
        XP, W3, b2, invpo, out);
}
